// AttentionTD_13280038879920
// MI455X (gfx1250) — matrix operand order repaired and checked by reading, not compiled, not run on hardware
//
#include <hip/hip_runtime.h>

typedef __attribute__((ext_vector_type(16))) __bf16 v16bf;
typedef __attribute__((ext_vector_type(8)))  __bf16 v8bf;
typedef __attribute__((ext_vector_type(2)))  __bf16 v2bf;
typedef __attribute__((ext_vector_type(8)))  float  v8f;
typedef __attribute__((ext_vector_type(8)))  unsigned int v8ui;

#define CIN    256
#define C1     128
#define HH     64
#define WW     64
#define HW     4096
#define NSAMP  256
#define NHEAD  4
#define HCH    32
#define RPEN   127

__device__ __forceinline__ __bf16 f2bf(float f) {
  unsigned int u = __builtin_bit_cast(unsigned int, f);
  unsigned int r = u + 0x7fffu + ((u >> 16) & 1u);   // round-to-nearest-even
  return __builtin_bit_cast(__bf16, (unsigned short)(r >> 16));
}

// Convert float pair -> packed {lo=bf16(a), hi=bf16(b)} dword.
__device__ __forceinline__ unsigned int f2bf2u(float a, float b) {
#if __has_builtin(__builtin_amdgcn_cvt_pk_bf16_f32)
  v2bf t = __builtin_amdgcn_cvt_pk_bf16_f32(a, b);
  return __builtin_bit_cast(unsigned int, t);
#else
  unsigned int ua = __builtin_bit_cast(unsigned int, a);
  unsigned int ub = __builtin_bit_cast(unsigned int, b);
  ua += 0x7fffu + ((ua >> 16) & 1u);
  ub += 0x7fffu + ((ub >> 16) & 1u);
  // result bytes: {ub[3],ub[2],ua[3],ua[2]}  (S0=ub -> sel 4..7, S1=ua -> sel 0..3)
  return __builtin_amdgcn_perm(ub, ua, 0x07060302u);
#endif
}

// ---------------------------------------------------------------------------
// 128x128 weight 1x1-conv GEMM:  Y[b][oc][s] = sum_c W[oc][c]*X[b][c][s]+bias[oc]
// Block: 256 threads (8 waves). Block tile: 128 oc x 64 s; wave: 16 oc x 64 s.
// X tile staged once per block in LDS, transposed [s][c], bf16 (packed c-pairs).
// grid = (S/64, B). ACC!=0 -> Y += result (compile-time; no branches in stores).
// ---------------------------------------------------------------------------
template <int ACC>
__global__ __launch_bounds__(256) void gemm128_wmma(
    const float* __restrict__ W, const float* __restrict__ bias,
    const float* __restrict__ X, long xBS,
    float* __restrict__ Y, long yBS, int S)
{
  __shared__ __bf16 sXT[64 * 32];           // [s_local][c_local], 4 KB
  unsigned int* sXTu = (unsigned int*)sXT;  // packed c-pair view: [s][c/2]
  const int tid  = threadIdx.x;
  const int b    = blockIdx.y;
  const int s0   = blockIdx.x * 64;
  const int wave = tid >> 5;
  const int lane = tid & 31;
  const int lh   = lane >> 4;
  const int lrow = lane & 15;
  const int ocB  = wave * 16;
  const int rp   = tid >> 4;                // staging: c-pair index 0..15
  const int c4   = tid & 15;                // staging: group of 4 s
  const float* Xb = X + (long)b * xBS;

  v8f accv[4] = {};
  for (int k0 = 0; k0 < 128; k0 += 32) {
    __syncthreads();                        // previous tile fully consumed
    // cooperative staging: each thread loads rows c=2rp,2rp+1 (float4 each),
    // packs pairs, stores 4 dwords. Prefetch next tile across the barrier.
    {
      const float4 va  = *(const float4*)(Xb + (long)(k0 + 2 * rp) * S + s0 + c4 * 4);
      const float4 vb4 = *(const float4*)(Xb + (long)(k0 + 2 * rp + 1) * S + s0 + c4 * 4);
      if (k0 + 32 < 128) {
        __builtin_prefetch(Xb + (long)(k0 + 32 + 2 * rp) * S + s0 + c4 * 4, 0, 3);
        __builtin_prefetch(Xb + (long)(k0 + 33 + 2 * rp) * S + s0 + c4 * 4, 0, 3);
      }
      sXTu[(c4 * 4 + 0) * 16 + rp] = f2bf2u(va.x, vb4.x);
      sXTu[(c4 * 4 + 1) * 16 + rp] = f2bf2u(va.y, vb4.y);
      sXTu[(c4 * 4 + 2) * 16 + rp] = f2bf2u(va.z, vb4.z);
      sXTu[(c4 * 4 + 3) * 16 + rp] = f2bf2u(va.w, vb4.w);
    }
    __syncthreads();

    // A fragment: W tile [16 oc x 32 c] -- one run of 16 consecutive floats
    const float* Wr = W + (ocB + lrow) * 128 + k0 + lh * 16;
    const float4 w0 = *(const float4*)(Wr);
    const float4 w1 = *(const float4*)(Wr + 4);
    const float4 w2 = *(const float4*)(Wr + 8);
    const float4 w3 = *(const float4*)(Wr + 12);
    v8ui ua;
    ua[0] = f2bf2u(w0.x, w0.y); ua[1] = f2bf2u(w0.z, w0.w);
    ua[2] = f2bf2u(w1.x, w1.y); ua[3] = f2bf2u(w1.z, w1.w);
    ua[4] = f2bf2u(w2.x, w2.y); ua[5] = f2bf2u(w2.z, w2.w);
    ua[6] = f2bf2u(w3.x, w3.y); ua[7] = f2bf2u(w3.z, w3.w);
    const v16bf a = __builtin_bit_cast(v16bf, ua);

#pragma unroll
    for (int nt = 0; nt < 4; ++nt) {
      const v16bf bm = *(const v16bf*)&sXT[(nt * 16 + lrow) * 32 + lh * 16];
      accv[nt] = __builtin_amdgcn_wmma_f32_16x16x32_bf16(
          false, a, false, bm, (short)0, accv[nt], false, false);
    }
  }

  // bias: the 8 values per lane are consecutive -> two b128 loads
  const float4 bz0 = *(const float4*)(bias + ocB + 8 * lh);
  const float4 bz1 = *(const float4*)(bias + ocB + 8 * lh + 4);
  const float be[8] = {bz0.x, bz0.y, bz0.z, bz0.w, bz1.x, bz1.y, bz1.z, bz1.w};
#pragma unroll
  for (int nt = 0; nt < 4; ++nt) {
#pragma unroll
    for (int e = 0; e < 8; ++e) {
      int oc = ocB + e + 8 * lh;
      long idx = (long)b * yBS + (long)oc * S + (s0 + nt * 16 + lrow);
      float val = accv[nt][e] + be[e];
      if (ACC) Y[idx] += val; else Y[idx] = val;
    }
  }
}

// ---------------------------------------------------------------------------
// Offset net: depthwise 4x4/s4 conv -> channel LN -> GELU -> 1x1 (128->2)
// -> pos = clip(offset + ref, -1, 1).  grid=(NSAMP,B), block=128 (1 thread/chan)
// ---------------------------------------------------------------------------
__global__ __launch_bounds__(128) void offset_net(
    const float* __restrict__ q, const float* __restrict__ dww,
    const float* __restrict__ dwb, const float* __restrict__ lng,
    const float* __restrict__ lnb, const float* __restrict__ pww,
    float* __restrict__ pos)
{
  const int b = blockIdx.y, n = blockIdx.x;
  const int hk = n >> 4, wk = n & 15;
  const int c = threadIdx.x;
  const float* qc = q + ((long)b * C1 + c) * HW;
  float s = dwb[c];
#pragma unroll
  for (int ky = 0; ky < 4; ++ky)
#pragma unroll
    for (int kx = 0; kx < 4; ++kx)
      s += dww[c * 16 + ky * 4 + kx] * qc[(hk * 4 + ky) * WW + wk * 4 + kx];

  __shared__ float red[128];
  __shared__ float red2[128];
  red[c] = s; __syncthreads();
  for (int st = 64; st; st >>= 1) { if (c < st) red[c] += red[c + st]; __syncthreads(); }
  float mu = red[0] * (1.f / 128.f); __syncthreads();
  float d = s - mu;
  red[c] = d * d; __syncthreads();
  for (int st = 64; st; st >>= 1) { if (c < st) red[c] += red[c + st]; __syncthreads(); }
  float var = red[0] * (1.f / 128.f); __syncthreads();
  float xln = d * rsqrtf(var + 1e-5f) * lng[c] + lnb[c];
  float g = 0.5f * xln * (1.f + erff(xln * 0.70710678118654752f));   // exact GELU
  red[c]  = g * pww[c];           // row 0 -> dy
  red2[c] = g * pww[128 + c];     // row 1 -> dx
  __syncthreads();
  for (int st = 64; st; st >>= 1) {
    if (c < st) { red[c] += red[c + st]; red2[c] += red2[c + st]; }
    __syncthreads();
  }
  if (c == 0) {
    float ry = ((float)hk + 0.5f) * (2.f / 15.f) - 1.f;
    float rx = ((float)wk + 0.5f) * (2.f / 15.f) - 1.f;
    float py = fminf(fmaxf(red[0]  + ry, -1.f), 1.f);
    float px = fminf(fmaxf(red2[0] + rx, -1.f), 1.f);
    pos[((long)b * NSAMP + n) * 2 + 0] = py;
    pos[((long)b * NSAMP + n) * 2 + 1] = px;
  }
}

// ---------------------------------------------------------------------------
// Bilinear-sample kv at deformed points: xs[b][c][n]. grid=(NSAMP,B), block=128.
// ---------------------------------------------------------------------------
__global__ __launch_bounds__(128) void deform_sample(
    const float* __restrict__ kv, const float* __restrict__ pos,
    float* __restrict__ xs)
{
  const int b = blockIdx.y, n = blockIdx.x, c = threadIdx.x;
  float py = pos[((long)b * NSAMP + n) * 2 + 0];
  float px = pos[((long)b * NSAMP + n) * 2 + 1];
  float X = (px + 1.f) * 0.5f * 63.f;
  float Y = (py + 1.f) * 0.5f * 63.f;
  float x0f = floorf(X), y0f = floorf(Y);
  float wx = X - x0f, wy = Y - y0f;
  int x0 = (int)x0f, y0 = (int)y0f;
  int x1c = min(x0 + 1, 63), y1c = min(y0 + 1, 63);
  int x0c = min(max(x0, 0), 63), y0c = min(max(y0, 0), 63);
  float w00 = (1.f - wx) * (1.f - wy), w10 = wx * (1.f - wy);
  float w01 = (1.f - wx) * wy,         w11 = wx * wy;
  const float* img = kv + ((long)b * CIN + c) * HW;
  float acc = w00 * img[y0c * WW + x0c] + w10 * img[y0c * WW + x1c]
            + w01 * img[y1c * WW + x0c] + w11 * img[y1c * WW + x1c];
  xs[((long)b * C1 + c) * NSAMP + n] = acc;
}

// ---------------------------------------------------------------------------
// Fused attention: scores(WMMA bf16) + RPE-bias(LDS bilinear) + softmax(shfl)
// + P*V (WMMA bf16). grid=(HW/128, NHEAD, B), block=256 (8 waves x 16 rows).
// sKT is [n][c] (transposed) so QK B-fragments are contiguous 32B LDS loads;
// sV is [c][n] so PV B-fragments are contiguous.
// ---------------------------------------------------------------------------
__global__ __launch_bounds__(256) void dat_attention(
    const float* __restrict__ q, const float* __restrict__ k,
    const float* __restrict__ v, const float* __restrict__ pos,
    const float* __restrict__ rpe, float* __restrict__ out)
{
  extern __shared__ char smem[];
  __bf16* sKT  = (__bf16*)smem;                     // [256][32]   16 KB
  __bf16* sV   = sKT + 32 * NSAMP;                  // [32][256]   16 KB
  __bf16* sP   = sV + 32 * NSAMP;                   // [8][16][256] 64 KB
  float*  sPos = (float*)(smem + 98304);            // [256][2]     2 KB
  float*  sRpe = sPos + 512;                        // [127*127]  ~63 KB

  const int b = blockIdx.z, h = blockIdx.y;
  const int tid = threadIdx.x, wave = tid >> 5, lane = tid & 31;
  const int lh = lane >> 4, lrow = lane & 15;

  const float* kb = k + ((long)b * C1 + h * HCH) * NSAMP;
  const float* vb = v + ((long)b * C1 + h * HCH) * NSAMP;
  // K: pack channel pairs -> sKT[n][c] dwords
  for (int i = tid; i < 4096; i += 256) {
    int c2 = i >> 8;                       // c = 2*c2, 2*c2+1
    int n  = i & 255;
    ((unsigned int*)sKT)[n * 16 + c2] = f2bf2u(kb[(2 * c2) * 256 + n],
                                               kb[(2 * c2 + 1) * 256 + n]);
  }
  // V: pack n pairs -> sV[c][n] dwords
  for (int i = tid; i < 4096; i += 256) {
    int c  = i >> 7;
    int n2 = i & 127;
    const float2 vv = *(const float2*)(vb + c * 256 + n2 * 2);
    ((unsigned int*)sV)[c * 128 + n2] = f2bf2u(vv.x, vv.y);
  }
  const float* pb = pos + (long)b * NSAMP * 2;
  for (int i = tid; i < 2 * NSAMP; i += 256) sPos[i] = pb[i];
  const float* rh = rpe + (long)h * RPEN * RPEN;
  for (int i = tid; i < RPEN * RPEN; i += 256) sRpe[i] = rh[i];
  __syncthreads();

  const int mBase = (blockIdx.x * 8 + wave) * 16;
  const float* qb = q + ((long)b * C1 + h * HCH) * HW;

  // A fragment of Q^T: [16 m x 32 c]; element pairs are consecutive channels
  v8ui uq;
#pragma unroll
  for (int t = 0; t < 8; ++t) {
    int kk = lh * 16 + t * 2;
    uq[t] = f2bf2u(qb[(long)kk * HW + mBase + lrow],
                   qb[(long)(kk + 1) * HW + mBase + lrow]);
  }
  const v16bf aq = __builtin_bit_cast(v16bf, uq);

  const float scale = 0.17677669529663689f;   // 32^-0.5
  v8f st[16];
#pragma unroll
  for (int nt = 0; nt < 16; ++nt) {
    const v16bf bk = *(const v16bf*)&sKT[(nt * 16 + lrow) * 32 + lh * 16];
    v8f c = {};
    c = __builtin_amdgcn_wmma_f32_16x16x32_bf16(false, aq, false, bk, (short)0, c, false, false);
    // add bilinear RPE bias per element
#pragma unroll
    for (int e = 0; e < 8; ++e) {
      int m = mBase + e + 8 * lh;
      int n = nt * 16 + lrow;
      float qy = (float)(m >> 6) * (2.f / 63.f) - 1.f;
      float qx = (float)(m & 63) * (2.f / 63.f) - 1.f;
      float dy = (qy - sPos[2 * n]) * 0.5f;
      float dx = (qx - sPos[2 * n + 1]) * 0.5f;
      float X = (dx + 1.f) * 63.f;              // (dx+1)*0.5*(127-1)
      float Y = (dy + 1.f) * 63.f;
      float x0f = floorf(X), y0f = floorf(Y);
      float wx = X - x0f, wy = Y - y0f;
      int x0 = (int)x0f, y0 = (int)y0f;
      int x1c = min(x0 + 1, RPEN - 1), y1c = min(y0 + 1, RPEN - 1);
      float w00 = (1.f - wx) * (1.f - wy), w10 = wx * (1.f - wy);
      float w01 = (1.f - wx) * wy,         w11 = wx * wy;
      float bz = w00 * sRpe[y0 * RPEN + x0]  + w10 * sRpe[y0 * RPEN + x1c]
               + w01 * sRpe[y1c * RPEN + x0] + w11 * sRpe[y1c * RPEN + x1c];
      c[e] = c[e] * scale + bz;
    }
    st[nt] = c;
  }

  // softmax over NS=256: rows live in 16-lane halves; reduce tiles then shfl_xor
#pragma unroll
  for (int e = 0; e < 8; ++e) {
    float mx = st[0][e];
#pragma unroll
    for (int nt = 1; nt < 16; ++nt) mx = fmaxf(mx, st[nt][e]);
#pragma unroll
    for (int o = 1; o < 16; o <<= 1) mx = fmaxf(mx, __shfl_xor(mx, o, 32));
    float sum = 0.f;
#pragma unroll
    for (int nt = 0; nt < 16; ++nt) { float p = __expf(st[nt][e] - mx); st[nt][e] = p; sum += p; }
#pragma unroll
    for (int o = 1; o < 16; o <<= 1) sum += __shfl_xor(sum, o, 32);
    float inv = 1.f / sum;
#pragma unroll
    for (int nt = 0; nt < 16; ++nt) st[nt][e] *= inv;
  }

  // stage P (C-layout -> LDS row-major) for A-layout reload; wave-private region,
  // DS pipeline is in-order within a wave.
  __bf16* sPw = sP + wave * 16 * NSAMP;
#pragma unroll
  for (int nt = 0; nt < 16; ++nt)
#pragma unroll
    for (int e = 0; e < 8; ++e)
      sPw[(e + 8 * lh) * NSAMP + nt * 16 + lrow] = f2bf(st[nt][e]);

  // out[m][c] = sum_n P[m,n] V[c,n]   (two 16-col c tiles, K=256 in 8 steps)
  v8f o0 = {}, o1 = {};
#pragma unroll
  for (int ks = 0; ks < 8; ++ks) {
    const __bf16* pA = &sPw[lrow * NSAMP + ks * 32 + lh * 16];
    const v8bf lo = *(const v8bf*)pA;            // kk = lh*16 + 0..7
    const v8bf hi = *(const v8bf*)(pA + 8);      // kk = lh*16 + 8..15
    const v16bf ap = __builtin_shufflevector(lo, hi,
        0, 1, 2, 3, 4, 5, 6, 7, 8, 9, 10, 11, 12, 13, 14, 15);
    const v16bf bv0 = *(const v16bf*)&sV[lrow * NSAMP + ks * 32 + lh * 16];
    const v16bf bv1 = *(const v16bf*)&sV[(16 + lrow) * NSAMP + ks * 32 + lh * 16];
    o0 = __builtin_amdgcn_wmma_f32_16x16x32_bf16(false, ap, false, bv0, (short)0, o0, false, false);
    o1 = __builtin_amdgcn_wmma_f32_16x16x32_bf16(false, ap, false, bv1, (short)0, o1, false, false);
  }
  // per-lane rows are 8 consecutive m -> two b128 stores per c-tile
  float* ob = out + ((long)b * C1 + h * HCH) * HW;
  float* p0 = ob + (long)lrow * HW + mBase + 8 * lh;
  float* p1 = ob + (long)(16 + lrow) * HW + mBase + 8 * lh;
  *(float4*)(p0)     = make_float4(o0[0], o0[1], o0[2], o0[3]);
  *(float4*)(p0 + 4) = make_float4(o0[4], o0[5], o0[6], o0[7]);
  *(float4*)(p1)     = make_float4(o1[0], o1[1], o1[2], o1[3]);
  *(float4*)(p1 + 4) = make_float4(o1[4], o1[5], o1[6], o1[7]);
}

// ---------------------------------------------------------------------------
// Initialize output: copy x0..x2 (float4), zero flow tail.
// ---------------------------------------------------------------------------
__global__ void init_out(const float4* __restrict__ x0, const float4* __restrict__ x1,
                         const float4* __restrict__ x2, float* __restrict__ out)
{
  long i = (long)blockIdx.x * blockDim.x + threadIdx.x;
  const long per4 = (long)4 * CIN * HW / 4;      // 1048576 float4s per variable
  float4* out4 = (float4*)out;
  if      (i <     per4) out4[i] = x0[i];
  else if (i < 2 * per4) out4[i] = x1[i - per4];
  else if (i < 3 * per4) out4[i] = x2[i - 2 * per4];
  else if (i < 3 * per4 + 12) out[3 * per4 * 4 + (i - 3 * per4)] = 0.f;  // flow
}

extern "C" void kernel_launch(void* const* d_in, const int* in_sizes, int n_in,
                              void* d_out, int out_size, void* d_ws, size_t ws_size,
                              hipStream_t stream) {
  (void)in_sizes; (void)n_in; (void)out_size; (void)ws_size;
  const float* x[3] = {(const float*)d_in[0], (const float*)d_in[1], (const float*)d_in[2]};
  const float* Wq  = (const float*)d_in[3];
  const float* bq  = (const float*)d_in[4];
  const float* Wk  = (const float*)d_in[5];
  const float* bk  = (const float*)d_in[6];
  const float* Wv  = (const float*)d_in[7];
  const float* bv  = (const float*)d_in[8];
  const float* Wo  = (const float*)d_in[9];
  const float* bo  = (const float*)d_in[10];
  const float* dww = (const float*)d_in[11];
  const float* dwb = (const float*)d_in[12];
  const float* lng = (const float*)d_in[13];
  const float* lnb = (const float*)d_in[14];
  const float* pww = (const float*)d_in[15];
  const float* rpe = (const float*)d_in[16];
  float* out = (float*)d_out;

  float* ws = (float*)d_ws;                 // ~18.4 MB used
  float* q_ws   = ws;                       // [4][128][4096]
  float* ao_ws  = q_ws  + 2097152;          // [4][128][4096]
  float* xs_ws  = ao_ws + 2097152;          // [4][128][256]
  float* k_ws   = xs_ws + 131072;
  float* v_ws   = k_ws  + 131072;
  float* pos_ws = v_ws  + 131072;           // [4][256][2]

  const long nInit = (long)3 * 4 * CIN * HW / 4 + 12;
  init_out<<<(unsigned)((nInit + 255) / 256), 256, 0, stream>>>(
      (const float4*)x[0], (const float4*)x[1], (const float4*)x[2], out);

  const int pi[3] = {1, 2, 2}, pj[3] = {0, 0, 1};
  const size_t shmem = 98304 + 2048 + RPEN * RPEN * sizeof(float);   // ~164.9 KB
  for (int p = 0; p < 3; ++p) {
    int i = pi[p], j = pj[p];
    // q = Wq @ x_i[:, :128] + bq
    gemm128_wmma<0><<<dim3(HW / 64, 4), 256, 0, stream>>>(
        Wq + p * 16384, bq + p * 128, x[i], (long)CIN * HW,
        q_ws, (long)C1 * HW, HW);
    offset_net<<<dim3(NSAMP, 4), 128, 0, stream>>>(
        q_ws, dww + p * 2048, dwb + p * 128, lng + p * 128, lnb + p * 128,
        pww + p * 256, pos_ws);
    deform_sample<<<dim3(NSAMP, 4), 128, 0, stream>>>(x[j], pos_ws, xs_ws);
    gemm128_wmma<0><<<dim3(NSAMP / 64, 4), 256, 0, stream>>>(
        Wk + p * 16384, bk + p * 128, xs_ws, (long)C1 * NSAMP,
        k_ws, (long)C1 * NSAMP, NSAMP);
    gemm128_wmma<0><<<dim3(NSAMP / 64, 4), 256, 0, stream>>>(
        Wv + p * 16384, bv + p * 128, xs_ws, (long)C1 * NSAMP,
        v_ws, (long)C1 * NSAMP, NSAMP);
    dat_attention<<<dim3(HW / 128, NHEAD, 4), 256, shmem, stream>>>(
        q_ws, k_ws, v_ws, pos_ws, rpe + (long)p * NHEAD * RPEN * RPEN, ao_ws);
    // res[i][:, 128:] += Wo @ attn_out + bo
    gemm128_wmma<1><<<dim3(HW / 64, 4), 256, 0, stream>>>(
        Wo + p * 16384, bo + p * 128, ao_ws, (long)C1 * HW,
        out + (long)i * 4 * CIN * HW + (long)C1 * HW, (long)CIN * HW, HW);
  }
}
